// density_learner_84482006712833
// MI455X (gfx1250) — compile-verified, matrix operand order checked by reading, not run on hardware
//
#include <hip/hip_runtime.h>
#include <stdint.h>

// ---- problem dims (fixed by reference) ----
#define BB 256
#define PP 1024
#define NN 4096
#define VV 512

static constexpr float ETA      = 0.001f / 0.03f;
static constexpr float V_STEP   = 0.01f;
static constexpr float U_FB     = 0.2f;
static constexpr float V_SPARSE = 0.1f;
static constexpr float B_DECAY  = 0.01f;

typedef __attribute__((ext_vector_type(16))) __bf16 v16bf;
typedef __attribute__((ext_vector_type(8)))  __bf16 v8bf;
typedef __attribute__((ext_vector_type(8)))  float  v8f;

__device__ __forceinline__ unsigned short f2bf(float f) {
  union { float f; uint32_t u; } x; x.f = f;
  uint32_t r = x.u + 0x7FFFu + ((x.u >> 16) & 1u);   // round-to-nearest-even
  return (unsigned short)(r >> 16);
}

__device__ __forceinline__ float thresh2(float u) {
  float pos = (u >  U_FB) ? (u - U_FB) : 0.0f;
  float neg = (u < -U_FB) ? (u + U_FB) : 0.0f;
  return pos + neg;
}

// ---- CDNA5 async global->LDS copy (ASYNCcnt path, bypasses VGPRs) ----
__device__ __forceinline__ void async_copy_b128(uint32_t lds_byte_addr, const void* gaddr) {
  asm volatile("global_load_async_to_lds_b128 %0, %1, off"
               :: "v"(lds_byte_addr), "v"((uint64_t)(uintptr_t)gaddr)
               : "memory");
}
__device__ __forceinline__ void wait_async0() {
  asm volatile("s_wait_asynccnt 0" ::: "memory");
}

// =====================================================================
// WMMA bf16 GEMM:  C[M,N] = A[M,K] * Bt[N,K]^T   (both operands K-major)
// 256 threads = 8 waves.  Tile: TM x 128 x 32, double-buffered LDS,
// async global->LDS loads overlap the WMMAs of the current tile.
//   TM=128: waves 2x4, wave tile 64x32 -> 4x2 wmma tiles (gram GEMM)
//   TM=64 : waves 1x8, wave tile 64x16 -> 4x1 wmma tiles (skinny-M GEMMs)
// EPI 0: store f32.  EPI 1: store bf16 of (acc - I)  (fused gram epilogue)
// =====================================================================
static constexpr int BNt = 128, BKt = 32, LSTR = 40; // 40*2B = 80B row stride (16B multiple)

template <int TM, int EPI>
__global__ __launch_bounds__(256) void gemm_bf16(
    const unsigned short* __restrict__ A,
    const unsigned short* __restrict__ Bt,
    void* __restrict__ C, int M, int N, int K)
{
  constexpr int WM  = (TM == 128) ? 2 : 1;       // wave grid M
  constexpr int WN  = 8 / WM;                    // wave grid N
  constexpr int FM  = (TM / WM) / 16;            // 4
  constexpr int FN  = (BNt / WN) / 16;           // 2 or 1
  constexpr int ACH = (TM * BKt / 8) / 256;      // A 16B-chunks per thread (2 or 1)
  constexpr int ABYTES = TM  * LSTR * 2;         // one A buffer, bytes
  constexpr int BBYTES = BNt * LSTR * 2;         // one B buffer, bytes

  __shared__ __align__(16) unsigned short As[2][TM  * LSTR];
  __shared__ __align__(16) unsigned short Bs[2][BNt * LSTR];

  const int tid    = threadIdx.x;
  const int lane   = tid & 31;
  const int wave   = tid >> 5;
  const int waveM  = wave / WN;
  const int waveN  = wave % WN;
  const int blockM = blockIdx.y * TM;
  const int blockN = blockIdx.x * BNt;
  const int lrow   = lane & 15;
  const int half   = (lane >> 4) * 8;   // lanes 0-15 -> K{0..7,16..23}; 16-31 -> K{8..15,24..31}

  const uint32_t asBase = (uint32_t)(uintptr_t)(&As[0][0]);
  const uint32_t bsBase = (uint32_t)(uintptr_t)(&Bs[0][0]);

  // issue one double-sided tile of async loads for k-offset kk into buffer buf
  auto load_tiles = [&](int kk, int buf) {
    uint32_t ab = asBase + (uint32_t)buf * ABYTES;
    uint32_t bb = bsBase + (uint32_t)buf * BBYTES;
#pragma unroll
    for (int i = 0; i < ACH; ++i) {
      int chunk = tid + i * 256;
      int row   = chunk >> 2;
      int cc    = (chunk & 3) * 8;
      async_copy_b128(ab + (uint32_t)(row * LSTR + cc) * 2,
                      &A[(size_t)(blockM + row) * K + kk + cc]);
    }
#pragma unroll
    for (int i = 0; i < 2; ++i) {
      int chunk = tid + i * 256;
      int row   = chunk >> 2;
      int cc    = (chunk & 3) * 8;
      async_copy_b128(bb + (uint32_t)(row * LSTR + cc) * 2,
                      &Bt[(size_t)(blockN + row) * K + kk + cc]);
    }
  };

  v8f acc[FM][FN] = {};

  const int nk = K / BKt;
  load_tiles(0, 0);                                  // prologue

  for (int it = 0; it < nk; ++it) {
    const int buf = it & 1;
    wait_async0();            // this wave's async loads into buf are complete
    __syncthreads();          // all waves' loads done; prior reads of buf^1 done
    if (it + 1 < nk) load_tiles((it + 1) * BKt, buf ^ 1);   // overlap with compute

    // ---- LDS -> fragments (two 16B ds loads each, per ISA 16-bit layout) ----
    v16bf afrag[FM], bfrag[FN];
#pragma unroll
    for (int fm = 0; fm < FM; ++fm) {
      const v8bf* pv = reinterpret_cast<const v8bf*>(
          &As[buf][(waveM * (TM / WM) + fm * 16 + lrow) * LSTR + half]);
      v8bf lo = pv[0], hi = pv[2];
      v16bf f;
#pragma unroll
      for (int e = 0; e < 8; ++e) { f[e] = lo[e]; f[e + 8] = hi[e]; }
      afrag[fm] = f;
    }
#pragma unroll
    for (int fn = 0; fn < FN; ++fn) {
      const v8bf* pv = reinterpret_cast<const v8bf*>(
          &Bs[buf][(waveN * (BNt / WN) + fn * 16 + lrow) * LSTR + half]);
      v8bf lo = pv[0], hi = pv[2];
      v16bf f;
#pragma unroll
      for (int e = 0; e < 8; ++e) { f[e] = lo[e]; f[e + 8] = hi[e]; }
      bfrag[fn] = f;
    }

#pragma unroll
    for (int fm = 0; fm < FM; ++fm)
#pragma unroll
      for (int fn = 0; fn < FN; ++fn)
        acc[fm][fn] = __builtin_amdgcn_wmma_f32_16x16x32_bf16(
            false, afrag[fm], false, bfrag[fn], (short)0, acc[fm][fn], false, false);
  }

  // ---- epilogue: C/D layout: VGPR i -> M = i + (lane>=16 ? 8 : 0), N = lane&15 ----
  const int mo = (lane >> 4) * 8;
#pragma unroll
  for (int fm = 0; fm < FM; ++fm) {
    int m0 = blockM + waveM * (TM / WM) + fm * 16 + mo;
#pragma unroll
    for (int fn = 0; fn < FN; ++fn) {
      int n = blockN + waveN * (BNt / WN) + fn * 16 + lrow;
#pragma unroll
      for (int i = 0; i < 8; ++i) {
        float val = acc[fm][fn][i];
        if constexpr (EPI == 1) {
          val -= ((m0 + i) == n) ? 1.0f : 0.0f;   // gram = phi^T phi - I
          reinterpret_cast<unsigned short*>(C)[(size_t)(m0 + i) * N + n] = f2bf(val);
        } else {
          reinterpret_cast<float*>(C)[(size_t)(m0 + i) * N + n] = val;
        }
      }
    }
  }
}

// =====================================================================
// Elementwise / reduction kernels
// =====================================================================
__global__ void init_kernel(float* u, float* v, unsigned short* vb, float* acc) {
  int i = blockIdx.x * 256 + threadIdx.x;
  if (i < BB * NN) u[i] = 0.0f;
  if (i < BB * VV) { v[i] = 0.1f; vb[i] = f2bf(0.1f); }
  if (i < 4) acc[i] = 0.0f;
}

__global__ void convert_kernel(const float* __restrict__ in, unsigned short* __restrict__ out, int n) {
  int i = blockIdx.x * 256 + threadIdx.x;
  if (i < n) out[i] = f2bf(in[i]);
}

// out[c*R + r] = in[r*C + c]  (transpose + f32->bf16), write-coalesced
__global__ void transpose_convert_kernel(const float* __restrict__ in, unsigned short* __restrict__ out,
                                         int R, int C) {
  int i = blockIdx.x * 256 + threadIdx.x;
  if (i < R * C) {
    int c = i / R, r = i - c * R;
    out[i] = f2bf(in[(size_t)r * C + c]);
  }
}

__global__ void thresh_store_kernel(const float* __restrict__ u, unsigned short* __restrict__ a) {
  int i = blockIdx.x * 256 + threadIdx.x;
  a[i] = f2bf(thresh2(u[i]));
}

__global__ void u_update_kernel(float* __restrict__ u, const float* __restrict__ drive,
                                const float* __restrict__ ag) {
  int i = blockIdx.x * 256 + threadIdx.x;
  float uu = u[i];
  u[i] = uu + ETA * (drive[i] - ag[i] - uu);
}

__global__ void err_kernel(const float* __restrict__ u, const float* __restrict__ logits,
                           unsigned short* __restrict__ err) {
  int i = blockIdx.x * 256 + threadIdx.x;
  float a = thresh2(u[i]);
  err[i] = f2bf(U_FB * (1.0f - a * __expf(-logits[i])));
}

__global__ void v_update_kernel(float* __restrict__ v, unsigned short* __restrict__ vb,
                                const float* __restrict__ dedv) {
  int i = blockIdx.x * 256 + threadIdx.x;
  float vv  = v[i];
  float sgn = (vv > 0.0f) ? 1.0f : ((vv < 0.0f) ? -1.0f : 0.0f);
  float nv  = vv - V_STEP * (dedv[i] + V_SPARSE * sgn);
  nv = (nv > 0.0f) ? nv : 0.0f;   // rectify_v
  v[i]  = nv;
  vb[i] = f2bf(nv);
}

__device__ __forceinline__ void block_sum_atomic(float val, float* dst) {
  __shared__ float sm[256];
  int t = threadIdx.x;
  sm[t] = val;
  __syncthreads();
#pragma unroll
  for (int s = 128; s > 0; s >>= 1) {
    if (t < s) sm[t] += sm[t + s];
    __syncthreads();
  }
  if (t == 0) atomicAdd(dst, sm[0]);
}

__global__ void loss_recon_kernel(const float* __restrict__ x, const float* __restrict__ xhat,
                                  float* acc) {
  int i = blockIdx.x * 256 + threadIdx.x;
  float d = x[i] - xhat[i];
  block_sum_atomic(d * d, acc);
}

__global__ void loss_fb_kernel(const float* __restrict__ u, const float* __restrict__ logits,
                               float* acc) {
  int i = blockIdx.x * 256 + threadIdx.x;
  float a = thresh2(u[i]);
  float l = logits[i];
  block_sum_atomic(a * __expf(-l) + l, acc);   // a/sigma + log(sigma)
}

__global__ void loss_abs_kernel(const float* __restrict__ p, float* acc) {
  int i = blockIdx.x * 256 + threadIdx.x;
  block_sum_atomic(fabsf(p[i]), acc);
}

__global__ void finalize_kernel(const float* __restrict__ acc, float* __restrict__ out) {
  out[0] = 0.5f * acc[0] / (float)BB      // recon
         + U_FB * acc[1] / (float)BB      // fb
         + V_SPARSE * acc[2] / (float)BB  // density
         + B_DECAY * acc[3] / (float)NN;  // b decay
}

// =====================================================================
// Host-side orchestration
// =====================================================================
extern "C" void kernel_launch(void* const* d_in, const int* in_sizes, int n_in,
                              void* d_out, int out_size, void* d_ws, size_t ws_size,
                              hipStream_t stream) {
  (void)in_sizes; (void)n_in; (void)out_size; (void)ws_size;
  const float* x   = (const float*)d_in[0];   // [B,P]
  const float* phi = (const float*)d_in[1];   // [P,N]
  const float* b   = (const float*)d_in[2];   // [N,V]
  float* out = (float*)d_out;

  // ---- carve workspace (256B aligned) ----
  char* w = (char*)d_ws;
  auto carve = [&](size_t bytes) { void* p = (void*)w; w += (bytes + 255) & ~(size_t)255; return p; };
  const size_t PN = (size_t)PP * NN, NV = (size_t)NN * VV, BP = (size_t)BB * PP,
               BN = (size_t)BB * NN, BVn = (size_t)BB * VV, NN2 = (size_t)NN * NN;

  unsigned short* phi_bf  = (unsigned short*)carve(PN  * 2);  // phi   [P,N]
  unsigned short* phiT_bf = (unsigned short*)carve(PN  * 2);  // phi^T [N,P]
  unsigned short* b_bf    = (unsigned short*)carve(NV  * 2);  // b     [N,V]
  unsigned short* bT_bf   = (unsigned short*)carve(NV  * 2);  // b^T   [V,N]
  unsigned short* x_bf    = (unsigned short*)carve(BP  * 2);  // x     [B,P]
  unsigned short* gram_bf = (unsigned short*)carve(NN2 * 2);  // gram  [N,N] (symmetric)
  float*          drive   = (float*)carve(BN  * 4);
  float*          u       = (float*)carve(BN  * 4);
  float*          v       = (float*)carve(BVn * 4);
  unsigned short* v_bf    = (unsigned short*)carve(BVn * 2);
  unsigned short* a_bf    = (unsigned short*)carve(BN  * 2);
  float*          ag      = (float*)carve(BN  * 4);
  float*          logits  = (float*)carve(BN  * 4);
  unsigned short* err_bf  = (unsigned short*)carve(BN  * 2);
  float*          dedv    = (float*)carve(BVn * 4);
  float*          xhat    = (float*)carve(BP  * 4);
  float*          acc     = (float*)carve(4 * 4);

  const int TB = 256;
  // ---- state + accumulator init (every call: must be deterministic) ----
  init_kernel<<<BN / TB, TB, 0, stream>>>(u, v, v_bf, acc);

  // ---- one-time operand conversions ----
  convert_kernel<<<(int)(PN / TB), TB, 0, stream>>>(phi, phi_bf, (int)PN);
  convert_kernel<<<(int)(NV / TB), TB, 0, stream>>>(b, b_bf, (int)NV);
  convert_kernel<<<(int)(BP / TB), TB, 0, stream>>>(x, x_bf, (int)BP);
  transpose_convert_kernel<<<(int)(PN / TB), TB, 0, stream>>>(phi, phiT_bf, PP, NN); // -> [N,P]
  transpose_convert_kernel<<<(int)(NV / TB), TB, 0, stream>>>(b, bT_bf, NN, VV);     // -> [V,N]

  // ---- gram = phi^T phi - I  (bf16 out, fused diag subtract), symmetric ----
  gemm_bf16<128, 1><<<dim3(NN / BNt, NN / 128), TB, 0, stream>>>(phiT_bf, phiT_bf, gram_bf, NN, NN, PP);
  // ---- drive = x @ phi ----
  gemm_bf16<64, 0><<<dim3(NN / BNt, BB / 64), TB, 0, stream>>>(x_bf, phiT_bf, drive, BB, NN, PP);

  // ---- 50 inference steps ----
  for (int s = 0; s < 50; ++s) {
    thresh_store_kernel<<<BN / TB, TB, 0, stream>>>(u, a_bf);
    // a @ gram (gram symmetric => its own transpose)
    gemm_bf16<64, 0><<<dim3(NN / BNt, BB / 64), TB, 0, stream>>>(a_bf, gram_bf, ag, BB, NN, NN);
    // logits = v @ b^T  (Bt operand = b natural layout), uses pre-update v
    gemm_bf16<64, 0><<<dim3(NN / BNt, BB / 64), TB, 0, stream>>>(v_bf, b_bf, logits, BB, NN, VV);
    u_update_kernel<<<BN / TB, TB, 0, stream>>>(u, drive, ag);
    err_kernel<<<BN / TB, TB, 0, stream>>>(u, logits, err_bf);
    // dedv = err @ b  (Bt operand = b^T)
    gemm_bf16<64, 0><<<dim3(VV / BNt, BB / 64), TB, 0, stream>>>(err_bf, bT_bf, dedv, BB, VV, NN);
    v_update_kernel<<<(int)(BVn / TB), TB, 0, stream>>>(v, v_bf, dedv);
  }

  // ---- outputs / loss ----
  thresh_store_kernel<<<BN / TB, TB, 0, stream>>>(u, a_bf);
  gemm_bf16<64, 0><<<dim3(NN / BNt, BB / 64), TB, 0, stream>>>(v_bf, b_bf, logits, BB, NN, VV); // final sigma
  gemm_bf16<64, 0><<<dim3(PP / BNt, BB / 64), TB, 0, stream>>>(a_bf, phi_bf, xhat, BB, PP, NN); // x_hat = a @ phi^T

  loss_recon_kernel<<<(int)(BP / TB), TB, 0, stream>>>(x, xhat, acc + 0);
  loss_fb_kernel<<<BN / TB, TB, 0, stream>>>(u, logits, acc + 1);
  loss_abs_kernel<<<(int)(BVn / TB), TB, 0, stream>>>(v, acc + 2);
  loss_abs_kernel<<<(int)(NV / TB), TB, 0, stream>>>(b, acc + 3);
  finalize_kernel<<<1, 1, 0, stream>>>(acc, out);
}
